// QuantizedLinear_16810501996913
// MI455X (gfx1250) — 8-bit matrix operand order repaired and checked by reading, compiled, not run on hardware
//
#include <hip/hip_runtime.h>
#include <stdint.h>

typedef int v8i __attribute__((ext_vector_type(8)));
typedef int v4iv __attribute__((ext_vector_type(4)));
typedef __attribute__((address_space(1))) v4iv* gp_v4i;
typedef __attribute__((address_space(3))) v4iv* lp_v4i;

#define TOKENS 8192
#define DIN    4096
#define DOUT   4096

#define TM 128
#define TN 128
#define TK 64
#define LDSS 80

#if defined(__has_builtin)
#if __has_builtin(__builtin_amdgcn_global_load_async_to_lds_b128)
#define HAVE_ASYNC_LDS 1
#endif
#endif
#ifndef HAVE_ASYNC_LDS
#define HAVE_ASYNC_LDS 0
#endif

__device__ __forceinline__ void async_copy16(const uint8_t* g, uint8_t* l) {
#if HAVE_ASYNC_LDS
    __builtin_amdgcn_global_load_async_to_lds_b128((gp_v4i)g, (lp_v4i)l, 0, 0);
#else
    *(int4*)l = *(const int4*)g;
#endif
}

__device__ __forceinline__ void wait_async0() {
#if defined(__has_builtin)
#if __has_builtin(__builtin_amdgcn_s_wait_asynccnt)
    __builtin_amdgcn_s_wait_asynccnt(0);
    return;
#endif
#endif
    asm volatile("s_wait_asynccnt 0" ::: "memory");
}

__device__ __forceinline__ int clampi(int v, int lo, int hi) {
    return v < lo ? lo : (v > hi ? hi : v);
}

__global__ __launch_bounds__(256) void quant_x_kernel(const float* __restrict__ x,
                                                      uint32_t* __restrict__ xq4,
                                                      int n4) {
    int i = blockIdx.x * 256 + threadIdx.x;
    if (i >= n4) return;
    float4 v = ((const float4*)x)[i];
    uint32_t q0 = (uint32_t)fminf(fmaxf(__builtin_rintf(v.x * 127.5f + 127.0f), 0.0f), 255.0f);
    uint32_t q1 = (uint32_t)fminf(fmaxf(__builtin_rintf(v.y * 127.5f + 127.0f), 0.0f), 255.0f);
    uint32_t q2 = (uint32_t)fminf(fmaxf(__builtin_rintf(v.z * 127.5f + 127.0f), 0.0f), 255.0f);
    uint32_t q3 = (uint32_t)fminf(fmaxf(__builtin_rintf(v.w * 127.5f + 127.0f), 0.0f), 255.0f);
    xq4[i] = q0 | (q1 << 8) | (q2 << 16) | (q3 << 24);
}

__global__ __launch_bounds__(256) void quant_w_kernel(const float* __restrict__ w,
                                                      uint32_t* __restrict__ wq4,
                                                      int* __restrict__ wsum) {
    __shared__ int red[256];
    const int o = blockIdx.x;
    const float* wr = w + (size_t)o * DIN;
    uint32_t* __restrict__ wqr = wq4 + (size_t)o * (DIN / 4);
    int sum = 0;
    for (int j = threadIdx.x; j < DIN / 4; j += 256) {
        float4 v = *(const float4*)(wr + j * 4);
        int a = clampi((int)__builtin_rintf(v.x), -128, 127);
        int b = clampi((int)__builtin_rintf(v.y), -128, 127);
        int c = clampi((int)__builtin_rintf(v.z), -128, 127);
        int d = clampi((int)__builtin_rintf(v.w), -128, 127);
        sum += a + b + c + d;
        wqr[j] = (uint32_t)(uint8_t)a | ((uint32_t)(uint8_t)b << 8) |
                 ((uint32_t)(uint8_t)c << 16) | ((uint32_t)(uint8_t)d << 24);
    }
    red[threadIdx.x] = sum;
    __syncthreads();
    for (int s = 128; s > 0; s >>= 1) {
        if (threadIdx.x < s) red[threadIdx.x] += red[threadIdx.x + s];
        __syncthreads();
    }
    if (threadIdx.x == 0) wsum[o] = red[0];
}

__global__ __launch_bounds__(256) void gemm_iu8_kernel(const uint8_t* __restrict__ xq,
                                                       const uint8_t* __restrict__ wq,
                                                       const int* __restrict__ wsum,
                                                       const float* __restrict__ bq,
                                                       float* __restrict__ out) {
    __shared__ __align__(16) uint8_t As[2][TM * LDSS];
    __shared__ __align__(16) uint8_t Bs[2][TN * LDSS];

    const int tid  = threadIdx.x;
    const int lane = tid & 31;
    const int wave = tid >> 5;
    const int wm   = wave >> 2;
    const int wn   = wave & 3;

    const int tBase = blockIdx.y * TM;
    const int oBase = blockIdx.x * TN;

    const int r0 = tid >> 2;
    const int kb = (tid & 3) * 16;

    const int h  = lane >> 4;
    const int lm = lane & 15;

    const uint8_t* gA0 = xq + (size_t)(tBase + r0) * DIN + kb;
    const uint8_t* gB0 = wq + (size_t)(oBase + r0) * DIN + kb;

    v8i acc[4][2];
#pragma unroll
    for (int mt = 0; mt < 4; ++mt)
#pragma unroll
        for (int nt = 0; nt < 2; ++nt)
            acc[mt][nt] = (v8i){};

    async_copy16(gA0,                      &As[0][(r0)      * LDSS + kb]);
    async_copy16(gA0 + (size_t)64 * DIN,   &As[0][(r0 + 64) * LDSS + kb]);
    async_copy16(gB0,                      &Bs[0][(r0)      * LDSS + kb]);
    async_copy16(gB0 + (size_t)64 * DIN,   &Bs[0][(r0 + 64) * LDSS + kb]);

    int cur = 0;
    for (int k = 0; k < DIN; k += TK) {
        wait_async0();
        __syncthreads();

        if (k + TK < DIN) {
            const int nx = cur ^ 1;
            async_copy16(gA0 + (k + TK),                    &As[nx][(r0)      * LDSS + kb]);
            async_copy16(gA0 + (size_t)64 * DIN + (k + TK), &As[nx][(r0 + 64) * LDSS + kb]);
            async_copy16(gB0 + (k + TK),                    &Bs[nx][(r0)      * LDSS + kb]);
            async_copy16(gB0 + (size_t)64 * DIN + (k + TK), &Bs[nx][(r0 + 64) * LDSS + kb]);
        }

        union { v8i v; uint64_t q[4]; } a[4];
#pragma unroll
        for (int mt = 0; mt < 4; ++mt) {
            const uint8_t* base = &As[cur][(wm * 64 + mt * 16 + lm) * LDSS + 16 * h];
            a[mt].q[0] = *(const uint64_t*)(base + 0);
            a[mt].q[1] = *(const uint64_t*)(base + 8);
            a[mt].q[2] = *(const uint64_t*)(base + 32);
            a[mt].q[3] = *(const uint64_t*)(base + 40);
        }
        union { v8i v; int4 p[2]; } b[2];
#pragma unroll
        for (int nt = 0; nt < 2; ++nt) {
            const uint8_t* base = &Bs[cur][(wn * 32 + nt * 16 + lm) * LDSS + 16 * h];
            b[nt].p[0] = *(const int4*)(base + 0);
            b[nt].p[1] = *(const int4*)(base + 32);
        }

#pragma unroll
        for (int mt = 0; mt < 4; ++mt)
#pragma unroll
            for (int nt = 0; nt < 2; ++nt)
                acc[mt][nt] = __builtin_amdgcn_wmma_i32_16x16x64_iu8(
                     false, a[mt].v,  true, b[nt].v,
                    acc[mt][nt],  false,  false);

        cur ^= 1;
    }

    const float S = (2.0f / 255.0f) * 0.01f;
#pragma unroll
    for (int nt = 0; nt < 2; ++nt) {
        const int   o    = oBase + wn * 32 + nt * 16 + lm;
        const int   wso  = wsum[o];
        const float bias = bq[o] * 1e-4f;
#pragma unroll
        for (int mt = 0; mt < 4; ++mt) {
            const int trow = tBase + wm * 64 + mt * 16 + 8 * h;
#pragma unroll
            for (int r = 0; r < 8; ++r) {
                float v = (float)(acc[mt][nt][r] - 127 * wso) * S + bias;
                out[(size_t)(trow + r) * DOUT + o] = v;
            }
        }
    }
}

extern "C" void kernel_launch(void* const* d_in, const int* in_sizes, int n_in,
                              void* d_out, int out_size, void* d_ws, size_t ws_size,
                              hipStream_t stream) {
    const float* x  = (const float*)d_in[0];
    const float* w  = (const float*)d_in[1];
    const float* bq = (const float*)d_in[2];
    float* out = (float*)d_out;

    uint8_t* ws   = (uint8_t*)d_ws;
    uint8_t* xq   = ws;
    uint8_t* wq   = ws + (size_t)TOKENS * DIN;
    int*     wsum = (int*)(ws + (size_t)TOKENS * DIN + (size_t)DOUT * DIN);

    const int n4 = TOKENS * DIN / 4;
    quant_x_kernel<<<(n4 + 255) / 256, 256, 0, stream>>>(x, (uint32_t*)xq, n4);
    quant_w_kernel<<<DOUT, 256, 0, stream>>>(w, (uint32_t*)wq, wsum);

    dim3 grid(DOUT / TN, TOKENS / TM);
    gemm_iu8_kernel<<<grid, 256, 0, stream>>>(xq, wq, wsum, bq, out);
}
